// GNN_61117384622111
// MI455X (gfx1250) — compile-verified, matrix operand order checked by reading, not run on hardware
//
#include <hip/hip_runtime.h>

// ---------------------------------------------------------------------------
// GCN pipeline for MI455X (gfx1250, wave32, WMMA)
//   h1 = relu(scatter(norm * (X@W1)[src], dst) + dinv^2*(X@W1) + b1)
//   h2 = relu(scatter(norm * (h1@W2)[src], dst) + dinv^2*(h1@W2) + b2)
//   pooled = segment_mean(h2, batch)
//   out = relu(pooled@Wf1 + bf1) @ Wf2 + bf2
// GEMMs: v_wmma_f32_16x16x32_bf16, fragments via ds_load_b128 pairs.
// B (weight) tiles staged with global_load_async_to_lds_b128 (ASYNCcnt path);
// weights pre-transposed to bf16 [N][K] so async chunks land layout-exact.
// ---------------------------------------------------------------------------

typedef __attribute__((ext_vector_type(16))) __bf16 v16bf;
typedef __attribute__((ext_vector_type(8)))  __bf16 v8bf;
typedef __attribute__((ext_vector_type(4)))  __bf16 v4bf;
typedef __attribute__((ext_vector_type(8)))  float  v8f;

#define HIDF 256      // hidden feature width (float)
#define HIDC 64       // hidden width in float4 chunks

#define BM 64
#define BN 128
#define BK 32
#define LDK 40        // padded K stride in LDS (80 bytes, 16B-aligned)

// ---------------- small utility kernels ----------------

__global__ __launch_bounds__(256) void fill_kernel(float* p, float v, int n) {
  int i = blockIdx.x * 256 + threadIdx.x;
  if (i < n) p[i] = v;
}

__global__ __launch_bounds__(256) void deg_kernel(const int* __restrict__ dst,
                                                  float* __restrict__ deg, int E) {
  int i = blockIdx.x * 256 + threadIdx.x;
  if (i < E) atomicAdd(&deg[dst[i]], 1.0f);
}

__global__ __launch_bounds__(256) void rsqrt_kernel(float* d, int n) {
  int i = blockIdx.x * 256 + threadIdx.x;
  if (i < n) d[i] = rsqrtf(d[i]);
}

// transpose + convert weights: W[K][N] f32  ->  Wt[N][K] bf16  (tiny, done once)
__global__ __launch_bounds__(256) void wt_convert_kernel(const float* __restrict__ W,
                                                         __bf16* __restrict__ Wt,
                                                         int K, int N) {
  int i = blockIdx.x * 256 + threadIdx.x;
  if (i >= N * K) return;
  int n = i / K, k = i - n * K;
  Wt[i] = (__bf16)W[(size_t)k * N + n];
}

// ---------------- WMMA bf16 GEMM: C = act(A[MxK] @ W[KxN] (+bias)) ----------
// 64x128 block tile, BK=32, 256 threads = 8 waves.
// Wave w: M-subtile (w&3), N-subtiles 4*(w>>2) .. +3  (A fragment reused x4).
// Requires: K % 32 == 0, N % 128 == 0 handled by grid, A 16B-aligned.

__global__ __launch_bounds__(256) void gemm_bf16_wmma(
    const float* __restrict__ A, const __bf16* __restrict__ Wt,  // Wt: [N][K]
    float* __restrict__ C, int M, int K, int N,
    const float* __restrict__ bias, int do_relu)
{
  __shared__ __bf16 As[BM][LDK];   // row  = M, K-contiguous
  __shared__ __bf16 Bs[BN][LDK];   // row  = N (W column), K-contiguous

  const int tid  = threadIdx.x;
  const int lane = tid & 31;
  const int wave = tid >> 5;          // 0..7
  const int mi   = wave & 3;          // M subtile
  const int ng   = (wave >> 2) * 4;   // first of 4 N subtiles

  const int bm = blockIdx.y * BM;
  const int bn = blockIdx.x * BN;

  v8f acc0 = {}, acc1 = {}, acc2 = {}, acc3 = {};

  const int half = lane >> 4;          // 0: lanes 0-15, 1: lanes 16-31
  const int l15  = lane & 15;
  const int arow = mi * 16 + l15;
  const int kb   = half * 8;           // per ISA 16-bit fragment K base

  for (int k0 = 0; k0 < K; k0 += BK) {
    // stage B tile (128 cols x 32 K, bf16, transposed): async global -> LDS,
    // 2 x 16B chunks per thread, no VGPR round trip, tracked by ASYNCcnt.
    #pragma unroll
    for (int it = 0; it < 2; ++it) {
      int q = tid + it * 256;          // 0..511
      int r = q >> 2;                  // 0..127 (column of W)
      int h = (q & 3) * 8;             // K sub-offset
      unsigned long long gaddr =
          (unsigned long long)(Wt + (size_t)(bn + r) * K + k0 + h);
      unsigned laddr = (unsigned)(uintptr_t)&Bs[r][h];  // low 32b = LDS offset
      asm volatile("global_load_async_to_lds_b128 %0, %1, off"
                   :: "v"(laddr), "v"(gaddr) : "memory");
    }
    // stage A tile (64x32 f32 -> bf16): 2 float4 chunks per thread, row guard only
    #pragma unroll
    for (int it = 0; it < 2; ++it) {
      int q = tid + it * 256;          // 0..511
      int r = q >> 3;
      int c = (q & 7) * 4;
      float4 v = make_float4(0.0f, 0.0f, 0.0f, 0.0f);
      int gr = bm + r;
      if (gr < M) v = *(const float4*)(A + (size_t)gr * K + k0 + c);
      v4bf b;
      b[0] = (__bf16)v.x; b[1] = (__bf16)v.y;
      b[2] = (__bf16)v.z; b[3] = (__bf16)v.w;
      *(v4bf*)&As[r][c] = b;
    }
    // all of this wave's async LDS writes done, then workgroup-wide fence
    asm volatile("s_wait_asynccnt 0x0" ::: "memory");
    __syncthreads();

    // fragments: two contiguous 16B runs each -> ds_load_b128
    v8bf alo = *(const v8bf*)&As[arow][kb];
    v8bf ahi = *(const v8bf*)&As[arow][kb + 16];
    v16bf af = __builtin_shufflevector(alo, ahi,
        0, 1, 2, 3, 4, 5, 6, 7, 8, 9, 10, 11, 12, 13, 14, 15);

    #pragma unroll
    for (int j = 0; j < 4; ++j) {
      int bcol = (ng + j) * 16 + l15;
      v8bf blo = *(const v8bf*)&Bs[bcol][kb];
      v8bf bhi = *(const v8bf*)&Bs[bcol][kb + 16];
      v16bf bfv = __builtin_shufflevector(blo, bhi,
          0, 1, 2, 3, 4, 5, 6, 7, 8, 9, 10, 11, 12, 13, 14, 15);
      v8f* accp = (j == 0) ? &acc0 : (j == 1) ? &acc1 : (j == 2) ? &acc2 : &acc3;
      *accp = __builtin_amdgcn_wmma_f32_16x16x32_bf16(
          false, af, false, bfv, (short)0, *accp, false, false);
    }
    __syncthreads();
  }

  // epilogue: C/D layout -> VGPR v holds row (v + 8*half), col = lane&15
  const int rbase = bm + mi * 16 + half * 8;
  #pragma unroll
  for (int j = 0; j < 4; ++j) {
    const v8f* accp = (j == 0) ? &acc0 : (j == 1) ? &acc1 : (j == 2) ? &acc2 : &acc3;
    int gcol = bn + (ng + j) * 16 + l15;
    float badd = bias ? bias[gcol] : 0.0f;
    #pragma unroll
    for (int v = 0; v < 8; ++v) {
      int row = rbase + v;
      if (row < M) {
        float xv = (*accp)[v] + badd;
        if (do_relu) xv = fmaxf(xv, 0.0f);
        C[(size_t)row * N + gcol] = xv;
      }
    }
  }
}

// ---------------- edge scatter: agg[dst] += dinv[src]*dinv[dst] * h[src] ----

__global__ __launch_bounds__(256) void scatter_kernel(
    const float* __restrict__ h, float* __restrict__ agg,
    const int* __restrict__ src, const int* __restrict__ dst,
    const float* __restrict__ dinv, int E)
{
  long long t = blockIdx.x * 256LL + threadIdx.x;
  if (t >= (long long)E * HIDC) return;
  int e = (int)(t >> 6);
  int c = (int)(t & (HIDC - 1));
  int s = src[e], d = dst[e];
  float nrm = dinv[s] * dinv[d];
  float4 v = ((const float4*)(h + (size_t)s * HIDF))[c];
  float* out = agg + (size_t)d * HIDF + c * 4;
  atomicAdd(out + 0, nrm * v.x);
  atomicAdd(out + 1, nrm * v.y);
  atomicAdd(out + 2, nrm * v.z);
  atomicAdd(out + 3, nrm * v.w);
}

// ---------------- fused self-loop + bias + relu (in place on agg) -----------

__global__ __launch_bounds__(256) void sl_bias_relu_kernel(
    float* __restrict__ agg, const float* __restrict__ h,
    const float* __restrict__ dinv, const float* __restrict__ bias, int Nn)
{
  long long t = blockIdx.x * 256LL + threadIdx.x;
  if (t >= (long long)Nn * HIDC) return;
  int i = (int)(t >> 6);
  int c = (int)(t & (HIDC - 1));
  float s = dinv[i] * dinv[i];            // self-loop norm = 1/deg
  float4 hv = ((const float4*)(h   + (size_t)i * HIDF))[c];
  float4 av = ((const float4*)(agg + (size_t)i * HIDF))[c];
  float4 bv = ((const float4*)bias)[c];
  float4 r;
  r.x = fmaxf(av.x + s * hv.x + bv.x, 0.0f);
  r.y = fmaxf(av.y + s * hv.y + bv.y, 0.0f);
  r.z = fmaxf(av.z + s * hv.z + bv.z, 0.0f);
  r.w = fmaxf(av.w + s * hv.w + bv.w, 0.0f);
  ((float4*)(agg + (size_t)i * HIDF))[c] = r;
}

// ---------------- pooling -----------------------------------------------

__global__ __launch_bounds__(256) void count_kernel(const int* __restrict__ batch,
                                                    float* __restrict__ cnt, int Nn) {
  int i = blockIdx.x * 256 + threadIdx.x;
  if (i < Nn) atomicAdd(&cnt[batch[i]], 1.0f);
}

__global__ __launch_bounds__(256) void pool_kernel(
    const float* __restrict__ a, const int* __restrict__ batch,
    float* __restrict__ psum, int Nn)
{
  long long t = blockIdx.x * 256LL + threadIdx.x;
  if (t >= (long long)Nn * HIDC) return;
  int i = (int)(t >> 6);
  int c = (int)(t & (HIDC - 1));
  int g = batch[i];
  float4 v = ((const float4*)(a + (size_t)i * HIDF))[c];
  float* out = psum + (size_t)g * HIDF + c * 4;
  atomicAdd(out + 0, v.x);
  atomicAdd(out + 1, v.y);
  atomicAdd(out + 2, v.z);
  atomicAdd(out + 3, v.w);
}

__global__ __launch_bounds__(256) void div_kernel(float* __restrict__ psum,
                                                  const float* __restrict__ cnt, int G)
{
  long long t = blockIdx.x * 256LL + threadIdx.x;
  if (t >= (long long)G * HIDC) return;
  int g = (int)(t >> 6);
  int c = (int)(t & (HIDC - 1));
  float inv = 1.0f / fmaxf(cnt[g], 1.0f);
  float4* p = (float4*)(psum + (size_t)g * HIDF) + c;
  float4 v = *p;
  v.x *= inv; v.y *= inv; v.z *= inv; v.w *= inv;
  *p = v;
}

// ---------------- final 128 -> 1 projection ------------------------------

__global__ __launch_bounds__(256) void final_kernel(
    const float* __restrict__ m1, const float* __restrict__ Wf2,
    const float* __restrict__ bf2, float* __restrict__ out, int G)
{
  int g = blockIdx.x * 256 + threadIdx.x;
  if (g >= G) return;
  const float4* row = (const float4*)(m1 + (size_t)g * 128);
  const float4* w   = (const float4*)Wf2;
  float acc = 0.0f;
  #pragma unroll 8
  for (int k = 0; k < 32; ++k) {
    float4 a = row[k], b = w[k];
    acc += a.x * b.x + a.y * b.y + a.z * b.z + a.w * b.w;
  }
  out[g] = acc + bf2[0];
}

// ---------------- host orchestration ------------------------------------

static inline int cdiv_ll(long long a, long long b) { return (int)((a + b - 1) / b); }

extern "C" void kernel_launch(void* const* d_in, const int* in_sizes, int n_in,
                              void* d_out, int out_size, void* d_ws, size_t ws_size,
                              hipStream_t stream) {
  const float* x   = (const float*)d_in[0];
  const int*   ei  = (const int*)d_in[1];
  const int*   bat = (const int*)d_in[2];
  const float* W1  = (const float*)d_in[3];
  const float* b1  = (const float*)d_in[4];
  const float* W2  = (const float*)d_in[5];
  const float* b2  = (const float*)d_in[6];
  const float* Wf1 = (const float*)d_in[7];
  const float* bf1 = (const float*)d_in[8];
  const float* Wf2 = (const float*)d_in[9];
  const float* bf2 = (const float*)d_in[10];

  const int E   = in_sizes[1] / 2;        // 1,600,000
  const int Nn  = in_sizes[2];            // 100,000
  const int IN  = in_sizes[0] / Nn;       // 128
  const int G   = out_size;               // 2048 (OUT_DIM = 1)
  const int F1  = in_sizes[8];            // 128 (MLP hidden)

  const int* src = ei;
  const int* dst = ei + E;

  // workspace layout
  float* bufA = (float*)d_ws;                      // [Nn, 256] GEMM output
  float* bufB = bufA + (size_t)Nn * HIDF;          // [Nn, 256] aggregation
  float* dinv = bufB + (size_t)Nn * HIDF;          // [Nn]
  float* psum = dinv + Nn;                         // [G, 256]
  float* cnt  = psum + (size_t)G * HIDF;           // [G]
  float* m1   = cnt + G;                           // [G, 128]
  __bf16* w1t  = (__bf16*)(m1 + (size_t)G * F1);   // [256, 128]  (Wt = W^T bf16)
  __bf16* w2t  = w1t + (size_t)HIDF * IN;          // [256, 256]
  __bf16* wf1t = w2t + (size_t)HIDF * HIDF;        // [128, 256]

  dim3 blk(256);

  // pre-transpose/convert all weights to bf16 [N][K]
  wt_convert_kernel<<<cdiv_ll((long long)IN * HIDF, 256), blk, 0, stream>>>(W1, w1t, IN, HIDF);
  wt_convert_kernel<<<cdiv_ll((long long)HIDF * HIDF, 256), blk, 0, stream>>>(W2, w2t, HIDF, HIDF);
  wt_convert_kernel<<<cdiv_ll((long long)HIDF * F1, 256), blk, 0, stream>>>(Wf1, wf1t, HIDF, F1);

  // degree / symmetric norm:  deg = 1 + in-degree ; dinv = rsqrt(deg)
  fill_kernel<<<cdiv_ll(Nn, 256), blk, 0, stream>>>(dinv, 1.0f, Nn);
  deg_kernel<<<cdiv_ll(E, 256), blk, 0, stream>>>(dst, dinv, E);
  rsqrt_kernel<<<cdiv_ll(Nn, 256), blk, 0, stream>>>(dinv, Nn);

  // layer 1: h1 = X @ W1
  {
    dim3 grid(HIDF / BN, cdiv_ll(Nn, BM));
    gemm_bf16_wmma<<<grid, blk, 0, stream>>>(x, w1t, bufA, Nn, IN, HIDF, nullptr, 0);
  }
  (void)hipMemsetAsync(bufB, 0, (size_t)Nn * HIDF * sizeof(float), stream);
  scatter_kernel<<<cdiv_ll((long long)E * HIDC, 256), blk, 0, stream>>>(
      bufA, bufB, src, dst, dinv, E);
  sl_bias_relu_kernel<<<cdiv_ll((long long)Nn * HIDC, 256), blk, 0, stream>>>(
      bufB, bufA, dinv, b1, Nn);

  // layer 2: h2 = a1 @ W2
  {
    dim3 grid(HIDF / BN, cdiv_ll(Nn, BM));
    gemm_bf16_wmma<<<grid, blk, 0, stream>>>(bufB, w2t, bufA, Nn, HIDF, HIDF, nullptr, 0);
  }
  (void)hipMemsetAsync(bufB, 0, (size_t)Nn * HIDF * sizeof(float), stream);
  scatter_kernel<<<cdiv_ll((long long)E * HIDC, 256), blk, 0, stream>>>(
      bufA, bufB, src, dst, dinv, E);
  sl_bias_relu_kernel<<<cdiv_ll((long long)Nn * HIDC, 256), blk, 0, stream>>>(
      bufB, bufA, dinv, b2, Nn);

  // global mean pool
  (void)hipMemsetAsync(psum, 0, (size_t)G * HIDF * sizeof(float), stream);
  (void)hipMemsetAsync(cnt, 0, (size_t)G * sizeof(float), stream);
  count_kernel<<<cdiv_ll(Nn, 256), blk, 0, stream>>>(bat, cnt, Nn);
  pool_kernel<<<cdiv_ll((long long)Nn * HIDC, 256), blk, 0, stream>>>(bufB, bat, psum, Nn);
  div_kernel<<<cdiv_ll((long long)G * HIDC, 256), blk, 0, stream>>>(psum, cnt, G);

  // MLP: m1 = relu(pooled @ Wf1 + bf1) ; out = m1 @ Wf2 + bf2
  {
    dim3 grid(F1 / BN, cdiv_ll(G, BM));
    gemm_bf16_wmma<<<grid, blk, 0, stream>>>(psum, wf1t, m1, G, HIDF, F1, bf1, 1);
  }
  final_kernel<<<cdiv_ll(G, 256), blk, 0, stream>>>(m1, Wf2, bf2, (float*)d_out, G);
}
